// DecoupledRadialAngularLoss_3100966388170
// MI455X (gfx1250) — matrix operand order repaired and checked by reading, not compiled, not run on hardware
//
#include <hip/hip_runtime.h>
#include <hip/hip_bf16.h>

// ---------------- constants ----------------
#define V_REAL   50257
#define V_PAD    50272          // multiple of 32 -> 32B-aligned bf16 rows in u_wT
#define N_SP     768            // spatial dims (K of the GEMM)
#define BL       2048           // B*L rows
#define R_MAX_C  3.0f
#define LAMBDA_R 0.1f
#define LOG_V_C  10.824907f     // ln(50257)
#define K_STEPS  (N_SP / 32)    // 24 WMMA K-steps
#define NB_STEPS ((V_REAL + 127) / 128)   // 393 column-block iterations
#define SPLIT    4                         // vocab-sweep split for occupancy
#define NB_CHUNK ((NB_STEPS + SPLIT - 1) / SPLIT)

typedef __bf16 bf16_t;
typedef __attribute__((ext_vector_type(16))) __bf16 v16bf;
typedef __attribute__((ext_vector_type(8)))  __bf16 v8bf;
typedef __attribute__((ext_vector_type(8)))  float  v8f;

union FragBF {
    v16bf v;
    v8bf  h[2];
};

__device__ __forceinline__ unsigned short f32_to_bf16_rne(float f) {
    unsigned int u = __float_as_uint(f);
    u += 0x7FFFu + ((u >> 16) & 1u);
    return (unsigned short)(u >> 16);
}

// ---------------- kernel 1: normalize vocab -> bf16, transposed [K][V_PAD] ----------------
// One block = 32 vocab rows. Pass 1: per-wave norm reduction. Pass 2: transposed
// coalesced bf16 stores (32 lanes -> 32 contiguous vocab columns).
__global__ void normalize_vocab_kernel(const float* __restrict__ W,
                                       unsigned short* __restrict__ uwT) {
    __shared__ float s_inv[32];
    const int wave = threadIdx.x >> 5;
    const int lane = threadIdx.x & 31;
    const int vbase = blockIdx.x * 32;

    for (int r = 0; r < 4; ++r) {
        const int v = vbase + wave * 4 + r;
        float ss = 0.0f;
        if (v < V_REAL) {
            const float* row = W + (size_t)v * (N_SP + 1) + 1;
            for (int k = lane; k < N_SP; k += 32) { float x = row[k]; ss += x * x; }
        }
        for (int off = 16; off; off >>= 1) ss += __shfl_xor(ss, off, 32);
        if (lane == 0)
            s_inv[wave * 4 + r] = (v < V_REAL) ? 1.0f / fmaxf(sqrtf(ss), 1e-12f) : 0.0f;
    }
    __syncthreads();

    const int v = vbase + lane;
    const bool valid = (v < V_REAL);
    const float inv = s_inv[lane];
    const float* row = W + (size_t)(valid ? v : 0) * (N_SP + 1) + 1;
    for (int k = wave; k < N_SP; k += 8) {
        float x = valid ? row[k] * inv : 0.0f;
        uwT[(size_t)k * V_PAD + vbase + lane] = f32_to_bf16_rne(x);
    }
}

// ---------------- kernel 2: normalize student -> bf16 [row][K]; radial partials ----------------
// Block = 8 rows; per-block radial partial {sum r_s, sum (r_s-r_t)^2, sum r_t, sum H}
// written deterministically to rad_part[blockIdx].
__global__ void normalize_student_kernel(const float* __restrict__ H,
                                         const float* __restrict__ TE,
                                         unsigned short* __restrict__ us,
                                         float* __restrict__ rad_part) {
    __shared__ float s_part[8][4];
    const int wave = threadIdx.x >> 5;
    const int lane = threadIdx.x & 31;
    const int row  = blockIdx.x * 8 + wave;

    const float* h = H + (size_t)row * (N_SP + 1);
    float ss = 0.0f;
    for (int k = lane; k < N_SP; k += 32) { float x = h[1 + k]; ss += x * x; }
    for (int off = 16; off; off >>= 1) ss += __shfl_xor(ss, off, 32);
    ss = __shfl(ss, 0, 32);
    const float inv = 1.0f / fmaxf(sqrtf(ss), 1e-12f);

    for (int k = lane; k < N_SP; k += 32)
        us[(size_t)row * N_SP + k] = f32_to_bf16_rne(h[1 + k] * inv);

    if (lane == 0) {
        float x0 = fmaxf(h[0], 1.0f + 1e-7f);
        float rs = logf(x0 + sqrtf(x0 * x0 - 1.0f));          // arccosh
        float hn = fminf(fmaxf(TE[row] / LOG_V_C, 0.0f), 1.0f);
        float rt = (1.0f / (1.0f + __expf(hn))) * R_MAX_C;    // sigmoid(-hn)*R_MAX
        float dr = rs - rt;
        s_part[wave][0] = rs;
        s_part[wave][1] = dr * dr;
        s_part[wave][2] = rt;
        s_part[wave][3] = hn;
    }
    __syncthreads();
    if (threadIdx.x < 4) {
        float a = 0.0f;
#pragma unroll
        for (int w = 0; w < 8; ++w) a += s_part[w][threadIdx.x];
        rad_part[blockIdx.x * 4 + threadIdx.x] = a;
    }
}

// ---------------- kernel 3: fused bf16 WMMA GEMM + online softmax KL ----------------
// Grid: 512 blocks x 256 threads (8 waves). Block owns 16 rows and 1/4 of the
// vocab sweep; wave w owns the 16x16 tile at columns nb*128 + w*16.
// Per-row partials {m, s, sum p*l, sum p*log p} go to row_part[split][row].
__global__ __launch_bounds__(256)
void kl_main_kernel(const unsigned short* __restrict__ us,
                    const unsigned short* __restrict__ uwT,
                    const float* __restrict__ p_teacher,
                    float* __restrict__ row_part) {
    __shared__ unsigned short ldsA[16 * N_SP];      // 24 KB
    __shared__ float red[8][16][4];                 // wave x row x {m,s,d,plogp}

    const int wave = threadIdx.x >> 5;
    const int lane = threadIdx.x & 31;
    const int row_base = (blockIdx.x >> 2) * 16;
    const int split    = blockIdx.x & 3;
    const int nb0 = split * NB_CHUNK;
    const int nb1 = (nb0 + NB_CHUNK < NB_STEPS) ? nb0 + NB_CHUNK : NB_STEPS;

    // cooperative load of the 16xK A-tile (bf16) into LDS
    {
        const unsigned int* src = (const unsigned int*)(us + (size_t)row_base * N_SP);
        unsigned int* dst = (unsigned int*)ldsA;
        for (int i = threadIdx.x; i < (16 * N_SP) / 2; i += 256) dst[i] = src[i];
    }
    __syncthreads();

    // per-lane online-softmax state for the 8 C-matrix row slots
    float m[8], s[8], d[8], pl[8];
#pragma unroll
    for (int i = 0; i < 8; ++i) { m[i] = -INFINITY; s[i] = 0.f; d[i] = 0.f; pl[i] = 0.f; }

    // A-fragment addressing: lane<16 -> K={k0..k0+7, k0+16..k0+23}; lane>=16 -> +8
    const int arow = lane & 15;
    const int ahalf = (lane >> 4) * 8;
    const unsigned short* aBase = &ldsA[arow * N_SP + ahalf];

    for (int nb = nb0; nb < nb1; ++nb) {
        const int col_base = nb * 128 + wave * 16;

        v8f accv = {};
#pragma unroll 4
        for (int ks = 0; ks < K_STEPS; ++ks) {
            const int k0 = ks * 32;
            FragBF a, b;
            a.h[0] = *(const v8bf*)(aBase + k0);
            a.h[1] = *(const v8bf*)(aBase + k0 + 16);
            // B-fragment: lane%16 = N column; K={k0..k0+7, k0+16..k0+23}, lane>=16 -> +8
            const unsigned short* bp = uwT + (size_t)(k0 + ahalf) * V_PAD + col_base + arow;
            __builtin_prefetch(bp + 128, 0, 3);     // next nb, same K-row, near-cache
#pragma unroll
            for (int s2 = 0; s2 < 8; ++s2) {
                b.v[s2]     = __builtin_bit_cast(bf16_t, bp[(size_t)s2 * V_PAD]);
                b.v[8 + s2] = __builtin_bit_cast(bf16_t, bp[(size_t)(16 + s2) * V_PAD]);
            }
            accv = __builtin_amdgcn_wmma_f32_16x16x32_bf16(
                false, a.v, false, b.v, (short)0, accv, false, false);
        }

        // C layout: VGPR i -> row (i or i+8), lane%16 -> column
        const int ncol = col_base + (lane & 15);
        const bool inb = (ncol < V_REAL);
        const int rbase = row_base + (lane >> 4) * 8;
        if (inb) {
#pragma unroll
            for (int i = 0; i < 8; ++i) {
                float l = accv[i];
                float p = p_teacher[(size_t)(rbase + i) * V_REAL + ncol];
                float mo = m[i];
                float mn = fmaxf(mo, l);
                s[i] = s[i] * __expf(mo - mn) + __expf(l - mn);
                m[i] = mn;
                d[i]  += p * l;
                pl[i] += p * __logf(fmaxf(p, 1e-12f));
            }
        }
    }

    // merge across the 16 lanes of each half-wave (LSE-merge for m,s; sums for d,pl)
#pragma unroll
    for (int off = 1; off < 16; off <<= 1) {
#pragma unroll
        for (int i = 0; i < 8; ++i) {
            float mo = __shfl_xor(m[i], off, 32);
            float so = __shfl_xor(s[i], off, 32);
            float mn = fmaxf(m[i], mo);
            s[i] = s[i] * __expf(m[i] - mn) + so * __expf(mo - mn);
            m[i] = mn;
            d[i]  += __shfl_xor(d[i],  off, 32);
            pl[i] += __shfl_xor(pl[i], off, 32);
        }
    }

    if ((lane & 15) == 0) {
        const int rr = (lane >> 4) * 8;
#pragma unroll
        for (int i = 0; i < 8; ++i) {
            red[wave][rr + i][0] = m[i];
            red[wave][rr + i][1] = s[i];
            red[wave][rr + i][2] = d[i];
            red[wave][rr + i][3] = pl[i];
        }
    }
    __syncthreads();

    // merge the 8 waves (disjoint column ranges of the same rows), write partial
    if (threadIdx.x < 16) {
        const int row = threadIdx.x;
        float M = -INFINITY, S = 0.f, D = 0.f, P = 0.f;
#pragma unroll
        for (int w = 0; w < 8; ++w) {
            float mw = red[w][row][0];
            float sw = red[w][row][1];
            float mn = fmaxf(M, mw);
            S = S * __expf(M - mn) + sw * __expf(mw - mn);
            M = mn;
            D += red[w][row][2];
            P += red[w][row][3];
        }
        float* q = row_part + ((size_t)split * BL + row_base + row) * 4;
        q[0] = M; q[1] = S; q[2] = D; q[3] = P;
    }
}

// ---------------- kernel 4: deterministic finalize ----------------
// One block, 256 threads. Merges SPLIT partials per row -> KL, tree-reduces all
// sums in a fixed order, writes the 6 outputs.
__global__ __launch_bounds__(256)
void finalize_kernel(const float* __restrict__ row_part,
                     const float* __restrict__ rad_part,
                     float* __restrict__ out) {
    __shared__ float red[256][5];
    const int tid = threadIdx.x;

    // KL: each thread owns rows tid, tid+256, ... (fixed order)
    float ksum = 0.0f;
    for (int r = tid; r < BL; r += 256) {
        float M = -INFINITY, S = 0.f, D = 0.f, P = 0.f;
#pragma unroll
        for (int sp = 0; sp < SPLIT; ++sp) {
            const float* q = row_part + ((size_t)sp * BL + r) * 4;
            float mw = q[0], sw = q[1];
            float mn = fmaxf(M, mw);
            S = S * __expf(M - mn) + sw * __expf(mw - mn);
            M = mn;
            D += q[2];
            P += q[3];
        }
        ksum += P - D + M + __logf(S);
    }

    // radial partials: thread tid owns block tid's partial (256 blocks)
    red[tid][0] = ksum;
    red[tid][1] = rad_part[tid * 4 + 0];
    red[tid][2] = rad_part[tid * 4 + 1];
    red[tid][3] = rad_part[tid * 4 + 2];
    red[tid][4] = rad_part[tid * 4 + 3];

    for (int off = 128; off; off >>= 1) {
        __syncthreads();
        if (tid < off) {
#pragma unroll
            for (int c = 0; c < 5; ++c) red[tid][c] += red[tid + off][c];
        }
    }
    __syncthreads();

    if (tid == 0) {
        const float invBL = 1.0f / (float)BL;
        float l_ang = red[0][0] * invBL;       // T=1 -> *T^2 identity
        float l_rad = red[0][2] * invBL;
        out[0] = l_ang + LAMBDA_R * l_rad;
        out[1] = l_ang;
        out[2] = l_rad;
        out[3] = red[0][1] * invBL;
        out[4] = red[0][3] * invBL;
        out[5] = red[0][4] * invBL;
    }
}

// ---------------- launcher ----------------
extern "C" void kernel_launch(void* const* d_in, const int* in_sizes, int n_in,
                              void* d_out, int out_size, void* d_ws, size_t ws_size,
                              hipStream_t stream) {
    (void)in_sizes; (void)n_in; (void)out_size; (void)ws_size;

    const float* h_student = (const float*)d_in[0];   // [2048][769]
    const float* W_vocab   = (const float*)d_in[1];   // [50257][769]
    const float* p_teacher = (const float*)d_in[2];   // [2048][50257]
    const float* t_entropy = (const float*)d_in[3];   // [2048]
    float* out = (float*)d_out;

    char* ws = (char*)d_ws;
    float* row_part = (float*)ws;                              // SPLIT*2048*4 floats (128 KB)
    float* rad_part = (float*)(ws + (size_t)SPLIT * BL * 4 * 4);               // 256*4 floats
    unsigned short* us  = (unsigned short*)(ws + 131072 + 4096);               // 3 MB bf16
    unsigned short* uwT = (unsigned short*)(ws + 131072 + 4096 + (size_t)BL * N_SP * 2); // 77 MB bf16

    normalize_vocab_kernel<<<V_PAD / 32, 256, 0, stream>>>(W_vocab, uwT);
    normalize_student_kernel<<<BL / 8, 256, 0, stream>>>(h_student, t_entropy, us, rad_part);
    kl_main_kernel<<<(BL / 16) * SPLIT, 256, 0, stream>>>(us, uwT, p_teacher, row_part);
    finalize_kernel<<<1, 256, 0, stream>>>(row_part, rad_part, out);
}
